// TransformerEncoderLayer_9758165696578
// MI455X (gfx1250) — matrix operand order repaired and checked by reading, not compiled, not run on hardware
//
#include <hip/hip_runtime.h>
#include <hip/hip_bf16.h>
#include <stdint.h>

// ---------------------------------------------------------------------------
// Transformer encoder layer for MI455X (gfx1250, wave32, WMMA).
// GEMMs in bf16 (V_WMMA_F32_16X16X32_BF16, f32 accumulate); softmax/LN in f32.
// DPP row reductions (pure VALU), async global->LDS double buffering
// (GLOBAL_LOAD_ASYNC_TO_LDS_B128 / ASYNCcnt), 64x64 wave tiles in the GEMM,
// barrier-free flash inner loop (wave-private P tile, LDS in-order per wave).
// ---------------------------------------------------------------------------

typedef __bf16 bf16;
typedef __attribute__((ext_vector_type(16))) __bf16 v16bf;
typedef __attribute__((ext_vector_type(8)))  float  v8f;

#define BB 4
#define SS 2048
#define DD 1024
#define HH 16
#define DH 64
#define DFF 4096
#define MM (BB * SS)          // 8192 rows

#if defined(__AMDGCN__) && __has_builtin(__builtin_amdgcn_global_load_async_to_lds_b128)
#define USE_ASYNC_LDS 1
#else
#define USE_ASYNC_LDS 0
#endif

__device__ __forceinline__ v8f v8zero() {
  v8f z = {0.f, 0.f, 0.f, 0.f, 0.f, 0.f, 0.f, 0.f};
  return z;
}

// Build a 16-element bf16 fragment from two 16-byte chunks.
__device__ __forceinline__ v16bf ld_frag(const bf16* p0, const bf16* p1) {
  union { struct { uint4 a, b; } u; v16bf v; } x;
  x.u.a = *(const uint4*)p0;
  x.u.b = *(const uint4*)p1;
  return x.v;
}

__device__ __forceinline__ v8f wmma_bf16(v16bf a, v16bf b, v8f c) {
  return __builtin_amdgcn_wmma_f32_16x16x32_bf16(false, a, false, b,
                                                 (short)0, c, false, false);
}

// ---- DPP cross-lane helpers (reduce within each 16-lane row, pure VALU) ----
template <int CTRL>
__device__ __forceinline__ float fdpp(float x) {
  return __int_as_float(
      __builtin_amdgcn_update_dpp(0, __float_as_int(x), CTRL, 0xF, 0xF, true));
}

// xor1 = quad_perm(1,0,3,2)=0xB1; xor2 = quad_perm(2,3,0,1)=0x4E;
// ROW_HALF_MIRROR=0x141 (i<->7-i); ROW_MIRROR=0x140 (i<->15-i).
__device__ __forceinline__ void red16_max(float (&v)[8]) {
#pragma unroll
  for (int j = 0; j < 8; ++j) v[j] = fmaxf(v[j], fdpp<0xB1>(v[j]));
#pragma unroll
  for (int j = 0; j < 8; ++j) v[j] = fmaxf(v[j], fdpp<0x4E>(v[j]));
#pragma unroll
  for (int j = 0; j < 8; ++j) v[j] = fmaxf(v[j], fdpp<0x141>(v[j]));
#pragma unroll
  for (int j = 0; j < 8; ++j) v[j] = fmaxf(v[j], fdpp<0x140>(v[j]));
}

__device__ __forceinline__ void red16_sum(float (&v)[8]) {
#pragma unroll
  for (int j = 0; j < 8; ++j) v[j] += fdpp<0xB1>(v[j]);
#pragma unroll
  for (int j = 0; j < 8; ++j) v[j] += fdpp<0x4E>(v[j]);
#pragma unroll
  for (int j = 0; j < 8; ++j) v[j] += fdpp<0x141>(v[j]);
#pragma unroll
  for (int j = 0; j < 8; ++j) v[j] += fdpp<0x140>(v[j]);
}

// ---- async global -> LDS copy of 16 bytes (CDNA5 path), sync fallback ----
__device__ __forceinline__ void cp16_to_lds(const bf16* g, bf16* l) {
#if USE_ASYNC_LDS
  typedef int v4i __attribute__((vector_size(16)));
  typedef __attribute__((address_space(1))) v4i* gp_t;
  typedef __attribute__((address_space(3))) v4i* lp_t;
  // AS3 pointer = low 32 bits of the generic LDS address (flat aperture rule)
  __builtin_amdgcn_global_load_async_to_lds_b128(
      (gp_t)(uintptr_t)g, (lp_t)(uint32_t)(uintptr_t)l, 0, 0);
#else
  *(uint4*)l = *(const uint4*)g;
#endif
}

__device__ __forceinline__ void wait_async_all() {
#if USE_ASYNC_LDS
  asm volatile("s_wait_asynccnt 0" ::: "memory");
#endif
}

// ---------------------------------------------------------------------------
// Packing / conversion kernels
// ---------------------------------------------------------------------------

__global__ __launch_bounds__(256) void k_cvt_f32_bf16(
    const float* __restrict__ in, bf16* __restrict__ out, long n) {
  long i = (long)blockIdx.x * 256 + threadIdx.x;
  if (i < n) out[i] = (bf16)in[i];
}

// Wq/Wk/Wv: [H, D, DH] f32 -> WqkvT: [3*H*DH][D] bf16 ; bqkv: [3*H*DH] f32
__global__ __launch_bounds__(256) void k_pack_wqkv(
    const float* __restrict__ Wq, const float* __restrict__ Wk,
    const float* __restrict__ Wv, const float* __restrict__ bq,
    const float* __restrict__ bk, const float* __restrict__ bv,
    bf16* __restrict__ WT, float* __restrict__ bvec) {
  long idx = (long)blockIdx.x * 256 + threadIdx.x;   // 3*H*DH*D total
  int d = (int)(idx % DD);
  long r = idx / DD;
  int e = (int)(r % DH); r /= DH;
  int h = (int)(r % HH);
  int p = (int)(r / HH);
  const float* W = (p == 0) ? Wq : (p == 1) ? Wk : Wv;
  float w = W[((long)h * DD + d) * DH + e];
  int n = p * (HH * DH) + h * DH + e;
  WT[(long)n * DD + d] = (bf16)w;
  if (d == 0) {
    const float* bb = (p == 0) ? bq : (p == 1) ? bk : bv;
    bvec[n] = bb[h * DH + e];
  }
}

// out[c*R + r] = (bf16) in[r*C + c]
__global__ __launch_bounds__(256) void k_transpose_bf16(
    const float* __restrict__ in, bf16* __restrict__ out, int R, int C) {
  long idx = (long)blockIdx.x * 256 + threadIdx.x;
  long r = idx / C;
  long c = idx % C;
  out[c * (long)R + r] = (bf16)in[idx];
}

// qkv [B*S][3072] (V at column 2048 + h*64 + e)  ->  vT [(b*H+h)*64+e][S]
__global__ __launch_bounds__(256) void k_pack_vT(
    const bf16* __restrict__ qkv, bf16* __restrict__ vT) {
  long idx = (long)blockIdx.x * 256 + threadIdx.x;   // B*H*DH*S
  int s = (int)(idx % SS);
  long r = idx / SS;
  int e = (int)(r % DH); r /= DH;
  int h = (int)(r % HH);
  int b = (int)(r / HH);
  vT[idx] = qkv[((long)(b * SS + s)) * (3 * DD) + 2 * DD + h * DH + e];
}

// ---------------------------------------------------------------------------
// Generic bf16 GEMM: C[M,N] = A[M,K] * B[K,N] + bias (+resid) (ReLU?)
// A row-major bf16, B given TRANSPOSED as BT[N][K] bf16.
// Block: 256 threads (8 waves as 2x4), tile 128x256, BK=32, double-buffered
// LDS filled with async global->LDS copies. Wave tile 64x64 (16 WMMAs/step).
// ---------------------------------------------------------------------------
template <bool RELU, bool RESID, bool OUT_BF16>
__global__ __launch_bounds__(256) void k_gemm_bf16(
    const bf16* __restrict__ A, const bf16* __restrict__ BT,
    const float* __restrict__ bias, const float* __restrict__ resid,
    float* __restrict__ outf, bf16* __restrict__ outb,
    int M, int N, int K) {
  const int tid   = threadIdx.x;
  const int wave  = tid >> 5;
  const int lane  = tid & 31;
  const int l16   = lane & 15;
  const int kh    = lane >> 4;
  const int waveM = wave & 1;   // 0..1 -> 64-row slice
  const int waveN = wave >> 1;  // 0..3 -> 64-col slice

  const int rowBase = blockIdx.x * 128;
  const int colBase = blockIdx.y * 256;

  __shared__ __align__(16) bf16 As[2][128 * 40];  // +8 pad: 80B row stride
  __shared__ __align__(16) bf16 Bs[2][256 * 40];

  v8f acc[4][4];
#pragma unroll
  for (int m = 0; m < 4; ++m)
#pragma unroll
    for (int n = 0; n < 4; ++n) acc[m][n] = v8zero();

  // A staging: thread t -> row t/2, 16 elems (2 x b128)
  const int aRow  = tid >> 1;
  const int aOff  = aRow * 40 + (tid & 1) * 16;
  const bf16* gA  = A + (long)(rowBase + aRow) * K + (tid & 1) * 16;
  // B staging: thread t -> row t, 32 elems (4 x b128)
  const int bOff  = tid * 40;
  const bf16* gB  = BT + (long)(colBase + tid) * K;

  const int niter = K >> 5;
  // prologue: stage tile 0 into buffer 0
  cp16_to_lds(gA, &As[0][aOff]);
  cp16_to_lds(gA + 8, &As[0][aOff + 8]);
#pragma unroll
  for (int c = 0; c < 4; ++c) cp16_to_lds(gB + c * 8, &Bs[0][bOff + c * 8]);

  for (int it = 0; it < niter; ++it) {
    const int cur = it & 1;
    wait_async_all();     // own async copies for buffer `cur` complete
    __syncthreads();      // whole tile visible to all waves
    if (it + 1 < niter) { // prefetch next tile into the other buffer
      const int ko = (it + 1) * 32;
      cp16_to_lds(gA + ko, &As[cur ^ 1][aOff]);
      cp16_to_lds(gA + ko + 8, &As[cur ^ 1][aOff + 8]);
#pragma unroll
      for (int c = 0; c < 4; ++c)
        cp16_to_lds(gB + ko + c * 8, &Bs[cur ^ 1][bOff + c * 8]);
    }

    v16bf afr[4], bfr[4];
#pragma unroll
    for (int m = 0; m < 4; ++m) {
      const bf16* p = As[cur] + (waveM * 64 + m * 16 + l16) * 40;
      afr[m] = ld_frag(p + kh * 8, p + 16 + kh * 8);
    }
#pragma unroll
    for (int n = 0; n < 4; ++n) {
      const bf16* p = Bs[cur] + (waveN * 64 + n * 16 + l16) * 40 + kh * 8;
      bfr[n] = ld_frag(p, p + 16);
    }
#pragma unroll
    for (int m = 0; m < 4; ++m)
#pragma unroll
      for (int n = 0; n < 4; ++n)
        acc[m][n] = wmma_bf16(afr[m], bfr[n], acc[m][n]);
  }

  // Epilogue. C layout: VGPR j -> row (j + 8*kh), col = l16 within 16x16 frag.
#pragma unroll
  for (int m = 0; m < 4; ++m) {
#pragma unroll
    for (int n = 0; n < 4; ++n) {
      const int col = colBase + waveN * 64 + n * 16 + l16;
      const float bv = bias[col];
#pragma unroll
      for (int j = 0; j < 8; ++j) {
        const int row = rowBase + waveM * 64 + m * 16 + kh * 8 + j;
        float v = acc[m][n][j] + bv;
        if (RESID) v += resid[(long)row * N + col];
        if (RELU) v = fmaxf(v, 0.f);
        if (OUT_BF16) outb[(long)row * N + col] = (bf16)v;
        else          outf[(long)row * N + col] = v;
      }
    }
  }
}

// ---------------------------------------------------------------------------
// Flash attention: one wave = 16 query rows, full DH=64 in registers.
// grid = (B*H, S/128); block = 256 threads (8 waves, fully independent).
// qkv: [B*S][3072] bf16 (q|k|v per head), vT: [(b*H+h)*64+e][S] bf16.
// ctx out: [B*S][D] bf16.
// ---------------------------------------------------------------------------
__global__ __launch_bounds__(256) void k_flash_attn(
    const bf16* __restrict__ qkv, const bf16* __restrict__ vT,
    bf16* __restrict__ ctx) {
  const int bh   = blockIdx.x;          // b*H + h
  const int b    = bh >> 4;
  const int h    = bh & 15;
  const int wave = threadIdx.x >> 5;
  const int lane = threadIdx.x & 31;
  const int l16  = lane & 15;
  const int kh   = lane >> 4;
  const int sRow = blockIdx.y * 128 + wave * 16;

  __shared__ __align__(16) bf16 pl[8][16 * 32];  // per-wave (private) P tile

  // Q fragments (A: 16x32 over DH split into two k-chunks)
  const bf16* qrow = qkv + ((long)(b * SS + sRow + l16)) * (3 * DD) + h * DH;
  v16bf qf0 = ld_frag(qrow + kh * 8,      qrow + 16 + kh * 8);
  v16bf qf1 = ld_frag(qrow + 32 + kh * 8, qrow + 48 + kh * 8);

  v8f acc[4];
#pragma unroll
  for (int nf = 0; nf < 4; ++nf) acc[nf] = v8zero();

  float mrun[8], lrun[8];
#pragma unroll
  for (int j = 0; j < 8; ++j) { mrun[j] = -__builtin_inff(); lrun[j] = 0.f; }

  const bf16* kbase = qkv + (long)b * SS * (3 * DD) + DD + h * DH;
  const bf16* vbase = vT + (long)bh * DH * SS;
  bf16* pw = pl[wave];

  for (int kb = 0; kb < SS; kb += 32) {
    // ---- scores S = Q * K^T for 32 keys (2 frags of 16 keys) ----
    v8f sc[2];
#pragma unroll
    for (int t = 0; t < 2; ++t) {
      const bf16* krow = kbase + (long)(kb + t * 16 + l16) * (3 * DD);
      v16bf k0 = ld_frag(krow + kh * 8,      krow + 16 + kh * 8);
      v16bf k1 = ld_frag(krow + 32 + kh * 8, krow + 48 + kh * 8);
      v8f s = wmma_bf16(qf0, k0, v8zero());
      s = wmma_bf16(qf1, k1, s);
      sc[t] = s * 0.125f;  // 1/sqrt(64)
    }

    // ---- online softmax; rows live per 16-lane half, DPP row reductions ----
    float bm[8];
#pragma unroll
    for (int j = 0; j < 8; ++j) bm[j] = fmaxf(sc[0][j], sc[1][j]);
    red16_max(bm);

    float alpha[8], rs[8];
#pragma unroll
    for (int j = 0; j < 8; ++j) {
      float mn = fmaxf(mrun[j], bm[j]);
      alpha[j] = __expf(mrun[j] - mn);
      mrun[j] = mn;
      sc[0][j] = __expf(sc[0][j] - mn);
      sc[1][j] = __expf(sc[1][j] - mn);
      rs[j] = sc[0][j] + sc[1][j];
    }
    red16_sum(rs);
#pragma unroll
    for (int j = 0; j < 8; ++j) lrun[j] = lrun[j] * alpha[j] + rs[j];
#pragma unroll
    for (int nf = 0; nf < 4; ++nf)
#pragma unroll
      for (int j = 0; j < 8; ++j) acc[nf][j] *= alpha[j];

    // ---- P: C layout -> A layout via wave-private LDS tile ----
    // (LDS ops are in-order per wave; no cross-wave sharing -> no barrier)
#pragma unroll
    for (int t = 0; t < 2; ++t)
#pragma unroll
      for (int j = 0; j < 8; ++j)
        pw[(j + 8 * kh) * 32 + t * 16 + l16] = (bf16)sc[t][j];

    const bf16* pr = pw + l16 * 32;
    v16bf pf = ld_frag(pr + kh * 8, pr + 16 + kh * 8);

    // ---- O += P * V ----
#pragma unroll
    for (int nf = 0; nf < 4; ++nf) {
      const bf16* vrow = vbase + (long)(nf * 16 + l16) * SS + kb + kh * 8;
      v16bf vf = ld_frag(vrow, vrow + 16);
      acc[nf] = wmma_bf16(pf, vf, acc[nf]);
    }
  }

  // ---- normalize + store ctx (bf16, [B*S][D]) ----
#pragma unroll
  for (int nf = 0; nf < 4; ++nf) {
#pragma unroll
    for (int j = 0; j < 8; ++j) {
      const int row = sRow + kh * 8 + j;
      const int col = h * DH + nf * 16 + l16;
      ctx[((long)(b * SS + row)) * DD + col] = (bf16)(acc[nf][j] / lrun[j]);
    }
  }
}

// ---------------------------------------------------------------------------
// LayerNorm over D=1024; one block (256 threads) per row; f32 (+optional bf16)
// ---------------------------------------------------------------------------
__global__ __launch_bounds__(256) void k_layernorm(
    const float* __restrict__ x, const float* __restrict__ g,
    const float* __restrict__ beta, float* __restrict__ outf,
    bf16* __restrict__ outb) {
  const long row = blockIdx.x;
  const float* xr = x + row * DD;
  const int wave = threadIdx.x >> 5;
  const int lane = threadIdx.x & 31;

  float s = 0.f, s2 = 0.f;
  for (int i = threadIdx.x; i < DD; i += 256) {
    float v = xr[i];
    s += v; s2 += v * v;
  }
#pragma unroll
  for (int mk = 16; mk >= 1; mk >>= 1) {
    s  += __shfl_xor(s,  mk, 32);
    s2 += __shfl_xor(s2, mk, 32);
  }
  __shared__ float red[2][8];
  if (lane == 0) { red[0][wave] = s; red[1][wave] = s2; }
  __syncthreads();
  float ts = 0.f, ts2 = 0.f;
#pragma unroll
  for (int w = 0; w < 8; ++w) { ts += red[0][w]; ts2 += red[1][w]; }
  const float mu  = ts * (1.f / DD);
  const float var = ts2 * (1.f / DD) - mu * mu;
  const float inv = rsqrtf(var + 1e-5f);

  for (int i = threadIdx.x; i < DD; i += 256) {
    float v = (xr[i] - mu) * inv * g[i] + beta[i];
    outf[row * DD + i] = v;
    if (outb) outb[row * DD + i] = (bf16)v;
  }
}

// ---------------------------------------------------------------------------
// Host-side launch sequence
// ---------------------------------------------------------------------------
extern "C" void kernel_launch(void* const* d_in, const int* in_sizes, int n_in,
                              void* d_out, int out_size, void* d_ws,
                              size_t ws_size, hipStream_t stream) {
  (void)in_sizes; (void)n_in; (void)out_size; (void)ws_size;

  const float* src   = (const float*)d_in[0];
  const float* Wq    = (const float*)d_in[1];
  const float* bq    = (const float*)d_in[2];
  const float* Wk    = (const float*)d_in[3];
  const float* bk    = (const float*)d_in[4];
  const float* Wv    = (const float*)d_in[5];
  const float* bv    = (const float*)d_in[6];
  const float* Wo    = (const float*)d_in[7];
  const float* bo    = (const float*)d_in[8];
  const float* ln1g  = (const float*)d_in[9];
  const float* ln1b  = (const float*)d_in[10];
  const float* W1    = (const float*)d_in[11];
  const float* b1    = (const float*)d_in[12];
  const float* W2    = (const float*)d_in[13];
  const float* b2    = (const float*)d_in[14];
  const float* ln2g  = (const float*)d_in[15];
  const float* ln2b  = (const float*)d_in[16];
  float* out = (float*)d_out;

  // Workspace layout (all offsets 256B-aligned; total ~310 MB)
  uint8_t* w = (uint8_t*)d_ws;
  constexpr size_t OFF_SRCB  = 0;
  constexpr size_t OFF_WQKVT = OFF_SRCB  + (size_t)MM * DD * 2;        // src bf16
  constexpr size_t OFF_BQKV  = OFF_WQKVT + (size_t)3 * DD * DD * 2;    // [3072][1024]
  constexpr size_t OFF_WOT   = OFF_BQKV  + (size_t)3 * DD * 4;
  constexpr size_t OFF_W1T   = OFF_WOT   + (size_t)DD * DD * 2;
  constexpr size_t OFF_W2T   = OFF_W1T   + (size_t)DFF * DD * 2;
  constexpr size_t OFF_QKV   = OFF_W2T   + (size_t)DD * DFF * 2;
  constexpr size_t OFF_VT    = OFF_QKV   + (size_t)MM * 3 * DD * 2;
  constexpr size_t OFF_CTX   = OFF_VT    + (size_t)BB * HH * DH * SS * 2;
  constexpr size_t OFF_XF    = OFF_CTX   + (size_t)MM * DD * 2;
  constexpr size_t OFF_X1F   = OFF_XF    + (size_t)MM * DD * 4;
  constexpr size_t OFF_X1B   = OFF_X1F   + (size_t)MM * DD * 4;
  constexpr size_t OFF_HB    = OFF_X1B   + (size_t)MM * DD * 2;
  constexpr size_t OFF_YF    = OFF_HB    + (size_t)MM * DFF * 2;

  bf16*  srcb  = (bf16*)(w + OFF_SRCB);
  bf16*  wqkvT = (bf16*)(w + OFF_WQKVT);
  float* bqkv  = (float*)(w + OFF_BQKV);
  bf16*  woT   = (bf16*)(w + OFF_WOT);
  bf16*  w1T   = (bf16*)(w + OFF_W1T);
  bf16*  w2T   = (bf16*)(w + OFF_W2T);
  bf16*  qkvb  = (bf16*)(w + OFF_QKV);
  bf16*  vTb   = (bf16*)(w + OFF_VT);
  bf16*  ctxb  = (bf16*)(w + OFF_CTX);
  float* xf    = (float*)(w + OFF_XF);
  float* x1f   = (float*)(w + OFF_X1F);
  bf16*  x1b   = (bf16*)(w + OFF_X1B);
  bf16*  hb    = (bf16*)(w + OFF_HB);
  float* yf    = (float*)(w + OFF_YF);

  // 1. src -> bf16
  k_cvt_f32_bf16<<<(MM * DD) / 256, 256, 0, stream>>>(src, srcb, (long)MM * DD);
  // 2. pack QKV weights transposed + packed bias
  k_pack_wqkv<<<(3 * DD * DD) / 256, 256, 0, stream>>>(Wq, Wk, Wv, bq, bk, bv,
                                                       wqkvT, bqkv);
  // 3. transpose other weights to bf16
  k_transpose_bf16<<<(DD * DD) / 256, 256, 0, stream>>>(Wo, woT, DD, DD);
  k_transpose_bf16<<<(DD * DFF) / 256, 256, 0, stream>>>(W1, w1T, DD, DFF);
  k_transpose_bf16<<<(DFF * DD) / 256, 256, 0, stream>>>(W2, w2T, DFF, DD);

  // 4. QKV projection: [8192,1024] x [1024,3072] + bqkv -> bf16
  k_gemm_bf16<false, false, true><<<dim3(MM / 128, (3 * DD) / 256), 256, 0,
                                    stream>>>(srcb, wqkvT, bqkv, nullptr,
                                              nullptr, qkvb, MM, 3 * DD, DD);
  // 5. V transpose for flash attention
  k_pack_vT<<<((long)BB * HH * DH * SS) / 256, 256, 0, stream>>>(qkvb, vTb);

  // 6. flash attention -> ctx bf16 [B*S][D]
  k_flash_attn<<<dim3(BB * HH, SS / 128), 256, 0, stream>>>(qkvb, vTb, ctxb);

  // 7. output projection + bo + residual(src) -> xf (f32)
  k_gemm_bf16<false, true, false><<<dim3(MM / 128, DD / 256), 256, 0, stream>>>(
      ctxb, woT, bo, src, xf, nullptr, MM, DD, DD);

  // 8. LN1 -> x1 (f32 + bf16)
  k_layernorm<<<MM, 256, 0, stream>>>(xf, ln1g, ln1b, x1f, x1b);

  // 9. FFN1: relu(x1 @ W1 + b1) -> bf16
  k_gemm_bf16<true, false, true><<<dim3(MM / 128, DFF / 256), 256, 0, stream>>>(
      x1b, w1T, b1, nullptr, nullptr, hb, MM, DFF, DD);

  // 10. FFN2: h @ W2 + b2 + residual(x1) -> yf (f32)
  k_gemm_bf16<false, true, false><<<dim3(MM / 128, DD / 256), 256, 0, stream>>>(
      hb, w2T, b2, x1f, yf, nullptr, MM, DD, DFF);

  // 11. LN2 -> d_out (f32)
  k_layernorm<<<MM, 256, 0, stream>>>(yf, ln2g, ln2b, out, nullptr);
}
